// BitLinear_88854283419865
// MI455X (gfx1250) — 8-bit matrix operand order repaired and checked by reading, compiled, not run on hardware
//
#include <hip/hip_runtime.h>

#define M_DIM 8192
#define N_DIM 4096
#define K_DIM 4096

typedef int v8i __attribute__((ext_vector_type(8)));


__global__ void k_zero(float* s) {
    if (threadIdx.x == 0) {
        s[0] = 0.0f;
        ((unsigned*)s)[1] = 0u;
        ((unsigned*)s)[2] = 0u;
    }
}

__global__ void k_wabs_partial(const float* __restrict__ w, float* __restrict__ part) {
    __shared__ float red[256];
    const int t = threadIdx.x;
    const long base = (long)blockIdx.x * 4096 + t * 16;
    const float4* p = (const float4*)(w + base);
    float s = 0.0f;
    #pragma unroll
    for (int j = 0; j < 4; ++j) {
        float4 v = p[j];
        s += fabsf(v.x) + fabsf(v.y) + fabsf(v.z) + fabsf(v.w);
    }
    red[t] = s; __syncthreads();
    for (int off = 128; off; off >>= 1) {
        if (t < off) red[t] += red[t + off];
        __syncthreads();
    }
    if (t == 0) part[blockIdx.x] = red[0];
}

__global__ void k_wabs_final(const float* __restrict__ part, float* scal) {
    __shared__ float red[256];
    const int t = threadIdx.x;
    float s = 0.0f;
    #pragma unroll
    for (int j = 0; j < 16; ++j) s += part[t + 256 * j];
    red[t] = s; __syncthreads();
    for (int off = 128; off; off >>= 1) {
        if (t < off) red[t] += red[t + off];
        __syncthreads();
    }
    if (t == 0) scal[0] = red[0];
}

__global__ void k_xmax(const float* __restrict__ x, unsigned* scal) {
    __shared__ unsigned red[256];
    const int t = threadIdx.x;
    const long base = (long)blockIdx.x * 8192 + t * 32;
    const float4* p = (const float4*)(x + base);
    unsigned m = 0u;
    #pragma unroll
    for (int j = 0; j < 8; ++j) {
        float4 v = p[j];
        unsigned a = __float_as_uint(fabsf(v.x)); m = (a > m) ? a : m;
        a = __float_as_uint(fabsf(v.y)); m = (a > m) ? a : m;
        a = __float_as_uint(fabsf(v.z)); m = (a > m) ? a : m;
        a = __float_as_uint(fabsf(v.w)); m = (a > m) ? a : m;
    }
    red[t] = m; __syncthreads();
    for (int off = 128; off; off >>= 1) {
        if (t < off) { unsigned o = red[t + off]; if (o > red[t]) red[t] = o; }
        __syncthreads();
    }
    if (t == 0) atomicMax(&scal[2], red[0]);
}

__global__ void k_quant_w(const float* __restrict__ w, const float* __restrict__ scal,
                          signed char* __restrict__ wq, unsigned* cnt) {
    __shared__ unsigned red[256];
    const float thr = 0.7f * (scal[0] * (1.0f / (4096.0f * 4096.0f)));
    const int t = threadIdx.x;
    const long base = (long)blockIdx.x * 4096 + t * 16;
    const float4* p = (const float4*)(w + base);
    union { signed char c[16]; int4 v; } u;
    unsigned nz = 0;
    #pragma unroll
    for (int j = 0; j < 4; ++j) {
        float4 v = p[j];
        float f[4] = { v.x, v.y, v.z, v.w };
        #pragma unroll
        for (int e = 0; e < 4; ++e) {
            signed char s = 0;
            if (fabsf(f[e]) >= thr) { s = (f[e] > 0.0f) ? 1 : -1; ++nz; }
            u.c[j * 4 + e] = s;
        }
    }
    *(int4*)(wq + base) = u.v;
    red[t] = nz; __syncthreads();
    for (int off = 128; off; off >>= 1) {
        if (t < off) red[t] += red[t + off];
        __syncthreads();
    }
    if (t == 0) atomicAdd(cnt, red[0]);
}

__global__ void k_quant_x(const float* __restrict__ x, const unsigned* __restrict__ scal,
                          signed char* __restrict__ xq) {
    const float mx = __uint_as_float(scal[2]);
    const float sx = 127.0f / fmaxf(mx, 1e-12f);
    const int t = threadIdx.x;
    const long base = (long)blockIdx.x * 4096 + t * 16;
    const float4* p = (const float4*)(x + base);
    union { signed char c[16]; int4 v; } u;
    #pragma unroll
    for (int j = 0; j < 4; ++j) {
        float4 v = p[j];
        float f[4] = { v.x, v.y, v.z, v.w };
        #pragma unroll
        for (int e = 0; e < 4; ++e) {
            float r = rintf(f[e] * sx);
            r = fminf(127.0f, fmaxf(-127.0f, r));
            u.c[j * 4 + e] = (signed char)(int)r;
        }
    }
    *(int4*)(xq + base) = u.v;
}

#define LDS_PITCH 80

__global__ void __launch_bounds__(256)
k_gemm(const signed char* __restrict__ xq, const signed char* __restrict__ wq,
       const float* __restrict__ scal, float* __restrict__ out) {
    __shared__ signed char lsA[2][256 * LDS_PITCH];
    __shared__ signed char lsB[2][128 * LDS_PITCH];

    const int t    = threadIdx.x;
    const int wave = t >> 5;
    const int lane = t & 31;
    const int half = lane >> 4;
    const int lr   = lane & 15;
    const int wm   = wave >> 1;
    const int wn   = wave & 1;

    const int m_base = blockIdx.y * 256;
    const int n_base = blockIdx.x * 128;

    const int r0 = t >> 2;
    const int c0 = (t & 3) * 16;

    const signed char* ga = xq + (size_t)(m_base + r0) * K_DIM + c0;
    const signed char* gb = wq + (size_t)(n_base + r0) * K_DIM + c0;
    const size_t rstep = (size_t)64 * K_DIM;

    v8i acc[4][4];
    #pragma unroll
    for (int mt = 0; mt < 4; ++mt)
        #pragma unroll
        for (int nt = 0; nt < 4; ++nt) {
            v8i z = { 0, 0, 0, 0, 0, 0, 0, 0 };
            acc[mt][nt] = z;
        }

    {
        uint4 a0 = *(const uint4*)(ga);
        uint4 a1 = *(const uint4*)(ga + rstep);
        uint4 a2 = *(const uint4*)(ga + 2 * rstep);
        uint4 a3 = *(const uint4*)(ga + 3 * rstep);
        uint4 b0 = *(const uint4*)(gb);
        uint4 b1 = *(const uint4*)(gb + rstep);
        *(uint4*)&lsA[0][(r0      ) * LDS_PITCH + c0] = a0;
        *(uint4*)&lsA[0][(r0 +  64) * LDS_PITCH + c0] = a1;
        *(uint4*)&lsA[0][(r0 + 128) * LDS_PITCH + c0] = a2;
        *(uint4*)&lsA[0][(r0 + 192) * LDS_PITCH + c0] = a3;
        *(uint4*)&lsB[0][(r0      ) * LDS_PITCH + c0] = b0;
        *(uint4*)&lsB[0][(r0 +  64) * LDS_PITCH + c0] = b1;
    }
    __syncthreads();

    for (int kb = 0; kb < K_DIM; kb += 64) {
        const int  cur  = (kb >> 6) & 1;
        const int  nxt  = cur ^ 1;
        const bool more = (kb + 64 < K_DIM);

        uint4 pa0, pa1, pa2, pa3, pb0, pb1;
        if (more) {
            const signed char* na = ga + kb + 64;
            const signed char* nb = gb + kb + 64;
            pa0 = *(const uint4*)(na);
            pa1 = *(const uint4*)(na + rstep);
            pa2 = *(const uint4*)(na + 2 * rstep);
            pa3 = *(const uint4*)(na + 3 * rstep);
            pb0 = *(const uint4*)(nb);
            pb1 = *(const uint4*)(nb + rstep);
        }

        v8i afr[4];
        #pragma unroll
        for (int mt = 0; mt < 4; ++mt) {
            const int row = wm * 64 + mt * 16 + lr;
            const signed char* pa = &lsA[cur][row * LDS_PITCH + half * 16];
            unsigned long long q0 = *(const unsigned long long*)(pa);
            unsigned long long q1 = *(const unsigned long long*)(pa + 8);
            unsigned long long q2 = *(const unsigned long long*)(pa + 32);
            unsigned long long q3 = *(const unsigned long long*)(pa + 40);
            v8i a = { (int)q0, (int)(q0 >> 32), (int)q1, (int)(q1 >> 32),
                      (int)q2, (int)(q2 >> 32), (int)q3, (int)(q3 >> 32) };
            afr[mt] = a;
        }
        v8i bfr[4];
        #pragma unroll
        for (int nt = 0; nt < 4; ++nt) {
            const int col = wn * 64 + nt * 16 + lr;
            const signed char* pb = &lsB[cur][col * LDS_PITCH + half * 16];
            uint4 p0 = *(const uint4*)(pb);
            uint4 p1 = *(const uint4*)(pb + 32);
            v8i b = { (int)p0.x, (int)p0.y, (int)p0.z, (int)p0.w,
                      (int)p1.x, (int)p1.y, (int)p1.z, (int)p1.w };
            bfr[nt] = b;
        }

        #pragma unroll
        for (int mt = 0; mt < 4; ++mt)
            #pragma unroll
            for (int nt = 0; nt < 4; ++nt)
                acc[mt][nt] = __builtin_amdgcn_wmma_i32_16x16x64_iu8(
                     true, afr[mt],  true, bfr[nt],
                    acc[mt][nt],  false,  false);

        if (more) {
            *(uint4*)&lsA[nxt][(r0      ) * LDS_PITCH + c0] = pa0;
            *(uint4*)&lsA[nxt][(r0 +  64) * LDS_PITCH + c0] = pa1;
            *(uint4*)&lsA[nxt][(r0 + 128) * LDS_PITCH + c0] = pa2;
            *(uint4*)&lsA[nxt][(r0 + 192) * LDS_PITCH + c0] = pa3;
            *(uint4*)&lsB[nxt][(r0      ) * LDS_PITCH + c0] = pb0;
            *(uint4*)&lsB[nxt][(r0 +  64) * LDS_PITCH + c0] = pb1;
            __syncthreads();
        }
    }

    const float    sumabs = scal[0];
    const unsigned cntnz  = ((const unsigned*)scal)[1];
    const unsigned mxbits = ((const unsigned*)scal)[2];
    const float inv_elems = 1.0f / (4096.0f * 4096.0f);
    const float abs_mean  = sumabs * inv_elems;
    const float nz_mean   = (float)cntnz * inv_elems + 1e-8f;
    const float scale_w   = abs_mean / nz_mean;
    const float scale_x   = 127.0f / fmaxf(__uint_as_float(mxbits), 1e-12f);
    const float out_scale = scale_w / scale_x;

    #pragma unroll
    for (int mt = 0; mt < 4; ++mt) {
        #pragma unroll
        for (int nt = 0; nt < 4; ++nt) {
            const int col   = n_base + wn * 64 + nt * 16 + lr;
            const int rbase = m_base + wm * 64 + mt * 16 + half * 8;
            #pragma unroll
            for (int i = 0; i < 8; ++i)
                out[(size_t)(rbase + i) * N_DIM + col] =
                    (float)acc[mt][nt][i] * out_scale;
        }
    }
}

extern "C" void kernel_launch(void* const* d_in, const int* in_sizes, int n_in,
                              void* d_out, int out_size, void* d_ws, size_t ws_size,
                              hipStream_t stream) {
    const float* x = (const float*)d_in[0];
    const float* w = (const float*)d_in[1];
    float* out = (float*)d_out;

    char* ws = (char*)d_ws;
    float* scal        = (float*)ws;
    float* part        = (float*)(ws + 4096);
    signed char* xq    = (signed char*)(ws + 65536);
    signed char* wq    = (signed char*)(ws + 65536 + (size_t)M_DIM * K_DIM);

    k_zero<<<1, 32, 0, stream>>>(scal);
    k_wabs_partial<<<4096, 256, 0, stream>>>(w, part);
    k_xmax<<<4096, 256, 0, stream>>>(x, (unsigned*)scal);
    k_wabs_final<<<1, 256, 0, stream>>>(part, scal);
    k_quant_w<<<4096, 256, 0, stream>>>(w, scal, wq, ((unsigned*)scal) + 1);
    k_quant_x<<<8192, 256, 0, stream>>>(x, (const unsigned*)scal, xq);

    dim3 grid(N_DIM / 128, M_DIM / 256);
    k_gemm<<<grid, 256, 0, stream>>>(xq, wq, scal, out);
}
